// ArithmeticCompute_1881195676356
// MI455X (gfx1250) — matrix operand order repaired and checked by reading, not compiled, not run on hardware
//
#include <hip/hip_runtime.h>

// CDNA5 / gfx1250 fused kernel for modular-circle arithmetic.
//  - mul path: out = a_soft^T * W_p * b_soft  ->  T = A_soft @ W_p via
//    v_wmma_f32_16x16x32_f16 (K padded to 32, N tiled by 16), then a
//    row-wise dot with b_soft using ds_swizzle xor reductions.
//  - inputs staged LDS-side via Tensor Data Mover (tensor_load_to_lds),
//    one 1280B 1-D tile per operand per wave, waited with s_wait_tensorcnt.

typedef __attribute__((ext_vector_type(16))) _Float16 v16h;
typedef __attribute__((ext_vector_type(8)))  _Float16 v8h;
typedef __attribute__((ext_vector_type(8)))  float    v8f;
typedef __attribute__((ext_vector_type(4)))  unsigned int u32x4;
typedef __attribute__((ext_vector_type(8)))  unsigned int u32x8;

#define NPRIMES 10
#define TEMPR   1000.0f
#define TWO_PI  6.28318530717958647692f

static constexpr int PRIMES_C[NPRIMES] = {2, 3, 5, 7, 11, 13, 17, 19, 23, 29};

// gfx1250 ds_swizzle group-of-32 xor butterfly: offset = (xor<<10)|(or<<5)|and
#define SWZ_XOR(v, m) \
    __uint_as_float(__builtin_amdgcn_ds_swizzle(__float_as_uint(v), (((m) << 10) | 0x1F)))

// 1-D TDM copy: ndw dwords from global -> LDS.  All args must be wave-uniform
// (SGPR).  D# per CDNA5 ISA ch.8: group0 = {count, lds_addr, global_addr, type=2},
// group1 = {data_size=4B, tensor_dim0 = tile_dim0 = ndw, dim1 = 1}.
__device__ __forceinline__ void tdm_load_1d(unsigned lds_off,
                                            unsigned ga_lo, unsigned ga_hi,
                                            unsigned ndw)
{
    u32x4 g0;
    g0[0] = 1u;                                   // count=1, not restore, no gather
    g0[1] = lds_off;                              // lds_addr (bytes)
    g0[2] = ga_lo;                                // global_addr[31:0]
    g0[3] = (ga_hi & 0x01FFFFFFu) | (2u << 30);   // global_addr[56:32] | type=2

    u32x8 g1;
    g1[0] = 0x00020000u;                          // wg_mask=0, data_size=2 (4B)
    g1[1] = (ndw & 0xFFFFu) << 16;                // tensor_dim0[15:0]  @ bits 63:48
    g1[2] = (ndw >> 16) | (1u << 16);             // tensor_dim0[31:16], tensor_dim1=1
    g1[3] = (ndw & 0xFFFFu) << 16;                // tile_dim0 @ bits 127:112
    g1[4] = 1u;                                   // tile_dim1=1, tile_dim2=0
    g1[5] = ndw;                                  // tensor_dim0_stride[31:0]
    g1[6] = 0u;
    g1[7] = 0u;

    asm volatile("tensor_load_to_lds %0, %1" :: "s"(g0), "s"(g1) : "memory");
}

__global__ __launch_bounds__(256) void circle_arith_kernel(
    const float* __restrict__ A, const float* __restrict__ Bc,
    float* __restrict__ out, int P)
{
    // TDM staging: [wave][0=a,1=b][16 pos * 20 floats]
    __shared__ __align__(16) float stage[8][2][320];
    // soft[wave][0=a,1=b][pos-in-tile][residue padded to 40 halves = 80B rows]
    __shared__ __align__(16) _Float16 soft[8][2][16][40];
    // W_p fragments in WMMA B-operand layout: [combo][0=cos,1=sin][lane][16 halves]
    __shared__ __align__(32) _Float16 wlut[14][2][32][16];
    __shared__ float sums[8][2][16];
    __shared__ float tmpl[NPRIMES][32][2];   // (cos,sin)(2*pi*r/p)

    const int tid  = threadIdx.x;
    const int wave = tid >> 5;
    const int lane = tid & 31;
    const int half = lane >> 4;   // 0 -> handles a, 1 -> handles b
    const int lp   = lane & 15;   // position within wave tile / col index

    // ---- kick off TDM loads of this wave's input tiles (overlap with LUT init)
    {
        const long tilebase = (long)blockIdx.x * 128 + (long)wave * 16;
        unsigned long long ga = (unsigned long long)(const void*)(A  + tilebase * 20);
        unsigned long long gb = (unsigned long long)(const void*)(Bc + tilebase * 20);
        unsigned lds_a = (unsigned)(unsigned long long)&stage[wave][0][0];
        unsigned ga_lo = __builtin_amdgcn_readfirstlane((unsigned)ga);
        unsigned ga_hi = __builtin_amdgcn_readfirstlane((unsigned)(ga >> 32));
        unsigned gb_lo = __builtin_amdgcn_readfirstlane((unsigned)gb);
        unsigned gb_hi = __builtin_amdgcn_readfirstlane((unsigned)(gb >> 32));
        lds_a = __builtin_amdgcn_readfirstlane(lds_a);
        tdm_load_1d(lds_a,        ga_lo, ga_hi, 320u);
        tdm_load_1d(lds_a + 1280, gb_lo, gb_hi, 320u);
    }

    // ---- init per-prime circle templates ----
    for (int i = tid; i < NPRIMES * 32; i += 256) {
        int pi = i >> 5, r = i & 31;
        int p = PRIMES_C[pi];
        float cx = 0.f, sx = 0.f;
        if (r < p) __sincosf(TWO_PI * (float)r / (float)p, &sx, &cx);
        tmpl[pi][r][0] = cx;
        tmpl[pi][r][1] = sx;
    }
    // ---- init W_p fragments (B operand layout, wave32):
    // lane<16: col n = lane,    elems j -> K = j
    // lane>=16: col n = lane-16, elems j -> K = 16 + j
    for (int c = wave; c < 14; c += 8) {
        int pi, t;
        if (c < 6) { pi = c; t = 0; }
        else       { pi = 6 + ((c - 6) >> 1); t = (c - 6) & 1; }
        int p  = PRIMES_C[pi];
        int n  = lp + 16 * t;
        int kb = half ? 16 : 0;
        #pragma unroll
        for (int j = 0; j < 16; ++j) {
            int k = kb + j;
            float cx = 0.f, sx = 0.f;
            if (k < p && n < p) {
                int r = (k * n) % p;
                __sincosf(TWO_PI * (float)r / (float)p, &sx, &cx);
            }
            wlut[c][0][lane][j] = (_Float16)cx;
            wlut[c][1][lane][j] = (_Float16)sx;
        }
    }

    // ---- wait for this wave's TDM tiles, then pull per-lane values ----
    asm volatile("s_wait_tensorcnt 0x0" ::: "memory");

    const long pos = (long)blockIdx.x * 128 + wave * 16 + lp;
    float vals[20];
    {
        const float4* sp = (const float4*)&stage[wave][half][lp * 20];
        #pragma unroll
        for (int q = 0; q < 5; ++q) {
            float4 v = sp[q];
            vals[4*q+0] = v.x; vals[4*q+1] = v.y;
            vals[4*q+2] = v.z; vals[4*q+3] = v.w;
        }
    }
    __syncthreads();   // wlut/tmpl ready for all waves

    // ---- add (half==0) / sub (half==1) planes via lane^16 ds_swizzle ----
    {
        float res[20];
        #pragma unroll
        for (int i = 0; i < NPRIMES; ++i) {
            float c0 = vals[2*i], s0 = vals[2*i+1];
            float c1 = SWZ_XOR(c0, 16);
            float s1 = SWZ_XOR(s0, 16);
            float x, y;
            if (half == 0) { // (ca,sa)=(c0,s0), (cb,sb)=(c1,s1): a+b
                x = c0*c1 - s0*s1; y = s0*c1 + c0*s1;
            } else {         // (cb,sb)=(c0,s0), (ca,sa)=(c1,s1): a-b = a*conj(b)
                x = c0*c1 + s0*s1; y = s1*c0 - c1*s0;
            }
            res[2*i] = x; res[2*i+1] = y;
        }
        float4* op = (float4*)(out + (size_t)half * (size_t)P * 20 + pos * 20);
        #pragma unroll
        for (int q = 0; q < 5; ++q) {
            float4 v;
            v.x = res[4*q+0]; v.y = res[4*q+1];
            v.z = res[4*q+2]; v.w = res[4*q+3];
            op[q] = v;
        }
    }

    float* mulout = out + (size_t)2 * (size_t)P * 20;

    #pragma unroll
    for (int pi = 0; pi < NPRIMES; ++pi) {
        const int p = PRIMES_C[pi];
        __syncthreads();

        // ---- phase 1: softmax (unnormalized) into LDS, f16 ----
        {
            float c = vals[2*pi], s = vals[2*pi+1];
            float sims[29];
            float mx = -3.0e38f;
            #pragma unroll
            for (int r = 0; r < p; ++r) {
                float sim = c * tmpl[pi][r][0] + s * tmpl[pi][r][1];
                sims[r] = sim;
                mx = fmaxf(mx, sim);
            }
            float sum = 0.f;
            _Float16* dst = &soft[wave][half][lp][0];
            #pragma unroll
            for (int r = 0; r < p; ++r) {
                float e = __expf(TEMPR * (sims[r] - mx));  // <= 1, no overflow
                sum += e;
                dst[r] = (_Float16)e;
            }
            #pragma unroll
            for (int r = p; r < 32; ++r) dst[r] = (_Float16)0.f;
            sums[wave][half][lp] = sum;
        }
        __syncthreads();

        // ---- phase 2: A fragment (16x32 A, contiguous k order matching wlut) ----
        // lane<16:  row=lane,    elems 0-15 -> K 0-15
        // lane>=16: row=lane-16, elems 0-15 -> K 16-31
        v16h afrag;
        {
            const _Float16* ap = &soft[wave][0][lp][0];
            const int akb = half ? 16 : 0;
            v8h lo = *(const v8h*)(ap + akb);
            v8h hi = *(const v8h*)(ap + akb + 8);
            #pragma unroll
            for (int j = 0; j < 8; ++j) { afrag[j] = lo[j]; afrag[8+j] = hi[j]; }
        }

        // this lane writes only row m0 (= lp-th reduction); hoist its scale/addr
        const int m0 = (half ? 8 : 0) + lp;
        const float inv =
            __builtin_amdgcn_rcpf(sums[wave][0][m0] * sums[wave][1][m0]);
        float* wptr = mulout + ((long)blockIdx.x * 128 + wave * 16 + m0) * 20 + pi * 2;

        const int ntile = (p > 16) ? 2 : 1;
        float accx[8], accy[8];
        #pragma unroll
        for (int r = 0; r < 8; ++r) { accx[r] = 0.f; accy[r] = 0.f; }

        #pragma unroll
        for (int t = 0; t < 2; ++t) {
            if (t < ntile) {
                const int c = (pi < 6) ? pi : (6 + 2 * (pi - 6) + t);
                v16h bx = *(const v16h*)&wlut[c][0][lane][0];
                v16h by = *(const v16h*)&wlut[c][1][lane][0];
                v8f z = {};
                v8f dx = __builtin_amdgcn_wmma_f32_16x16x32_f16(
                    false, afrag, false, bx, (short)0, z, false, false);
                v8f dy = __builtin_amdgcn_wmma_f32_16x16x32_f16(
                    false, afrag, false, by, (short)0, z, false, false);
                // D layout: VGPR r, lanes 0-15 -> (M=r, N=lane); lanes 16-31 -> (M=r+8, N=lane-16)
                #pragma unroll
                for (int r = 0; r < 8; ++r) {
                    const int m  = (half ? 8 : 0) + r;
                    const int ng = lp + 16 * t;
                    float bs = (float)soft[wave][1][m][ng];
                    accx[r] = fmaf(dx[r], bs, accx[r]);
                    accy[r] = fmaf(dy[r], bs, accy[r]);
                }
            }
        }

        // ---- phase 3: 16-lane row reduction via ds_swizzle xor butterflies
        // (masks < 16 keep the two row-halves separate) ----
        #pragma unroll
        for (int r = 0; r < 8; ++r) {
            float vx = accx[r], vy = accy[r];
            vx += SWZ_XOR(vx, 1); vy += SWZ_XOR(vy, 1);
            vx += SWZ_XOR(vx, 2); vy += SWZ_XOR(vy, 2);
            vx += SWZ_XOR(vx, 4); vy += SWZ_XOR(vy, 4);
            vx += SWZ_XOR(vx, 8); vy += SWZ_XOR(vy, 8);
            if (lp == r) {   // this lane's designated row: m == m0
                float2 o; o.x = vx * inv; o.y = vy * inv;
                *(float2*)wptr = o;
            }
        }
    }
}

extern "C" void kernel_launch(void* const* d_in, const int* in_sizes, int n_in,
                              void* d_out, int out_size, void* d_ws, size_t ws_size,
                              hipStream_t stream) {
    const float* a = (const float*)d_in[0];   // [B,S,10,2] f32
    const float* b = (const float*)d_in[1];   // [B,S,10,2] f32
    float* out = (float*)d_out;               // [3,B,S,10,2] f32

    const int P = in_sizes[0] / (NPRIMES * 2);   // B*S = 65536 (multiple of 128)
    const int blocks = P / 128;                  // 512 blocks * 8 waves * 16 pos
    circle_arith_kernel<<<dim3(blocks), dim3(256), 0, stream>>>(a, b, out, P);
}
